// SparseLaneAttention_39006892982753
// MI455X (gfx1250) — matrix operand order repaired and checked by reading, not compiled, not run on hardware
//
#include <hip/hip_runtime.h>
#include <hip/hip_bf16.h>
#include <stdint.h>

// ---------------------------------------------------------------------------
// SparseLaneAttention for MI455X (gfx1250), wave32 + WMMA bf16 (f32 accum).
// All matmuls -> v_wmma_f32_16x16x32_bf16. Scatter -> global_atomic_add_f32.
// Memory->LDS staging uses GLOBAL_LOAD_ASYNC_TO_LDS_B128 (ASYNCcnt) when the
// toolchain exposes the builtin; intra-wave LDS transposes use s_wait_dscnt.
// ---------------------------------------------------------------------------

#define DD 128
static constexpr int Bm = 16, Nn = 1024, Mrows = Bm * Nn;   // 16384 rows
static constexpr int Ss = 6, Ll = 4, Ee = 32768;
static constexpr float NORMF = 0.08838834764831845f;        // 1/sqrt(128)
static constexpr float EPSF  = 1e-5f;

typedef __bf16 bf16_t;
typedef __attribute__((ext_vector_type(16))) __bf16 v16bf;
typedef __attribute__((ext_vector_type(8)))  float  v8f;
typedef __attribute__((ext_vector_type(4)))  int    v4i;

#if defined(__gfx1250__) && \
    __has_builtin(__builtin_amdgcn_global_load_async_to_lds_b128)
#define USE_ASYNC_LDS 1
#else
#define USE_ASYNC_LDS 0
#endif

#if USE_ASYNC_LDS
// Param types per hipcc diagnostic: (v4i __device__*, v4i __shared__*, Ii, Ii)
typedef __attribute__((address_space(1))) v4i gv4i;   // global int4
typedef __attribute__((address_space(3))) v4i lv4i;   // LDS int4
#define ASYNC_B128(gp, lp)                                                    \
  __builtin_amdgcn_global_load_async_to_lds_b128((gv4i*)(gp), (lv4i*)(lp), 0, 0)
#endif

__device__ __forceinline__ bf16_t f2bf(float f) {
  union { float f; uint32_t u; } a; a.f = f;
  uint32_t r = a.u + 0x7FFFu + ((a.u >> 16) & 1u);   // round-to-nearest-even
  uint16_t h = (uint16_t)(r >> 16);
  bf16_t b; __builtin_memcpy(&b, &h, 2); return b;
}

__device__ __forceinline__ v8f wmma_bf(v16bf a, v16bf b, v8f c) {
  // 8 args: (neg_a, A, neg_b, B, c_mod, C, reuse_a, reuse_b)
  return __builtin_amdgcn_wmma_f32_16x16x32_bf16(false, a, false, b,
                                                 (short)0, c, false, false);
}

__device__ __forceinline__ float waveSum(float s) {
#pragma unroll
  for (int msk = 16; msk; msk >>= 1) s += __shfl_xor(s, msk, 32);
  return s;
}

// ------------------------------- f32 -> bf16 -------------------------------
__global__ void k_f32_to_bf16(const float* __restrict__ s,
                              bf16_t* __restrict__ d, int n) {
  int i = blockIdx.x * blockDim.x + threadIdx.x;
  if (i < n) d[i] = f2bf(s[i]);
}

// ------------------------- y[M,128] = x[M,128] @ W^T -----------------------
// Block = 8 waves; block stages one 16x128 A tile in LDS (async-to-LDS when
// available); wave w does cols [16w,16w+16). nn.Linear: y[m,f] = sum_d
// x[m,d]*W[f,d] -> B columns are contiguous rows of W. Output mode is a
// compile-time template parameter so the epilogue is branch-free.
template <bool BF16OUT>
__global__ void __launch_bounds__(256)
k_gemm_xwT(const bf16_t* __restrict__ X, const bf16_t* __restrict__ W,
           void* __restrict__ out) {
  __shared__ bf16_t As[16 * DD];
  const int tid = threadIdx.x;
  const int rowBase = blockIdx.x * 16;
#if USE_ASYNC_LDS
  // 256 threads x one b128 = the whole 4KB tile, memory->LDS direct.
  ASYNC_B128(X + (size_t)rowBase * DD + tid * 8, As + tid * 8);
  asm volatile("s_wait_asynccnt 0x0" ::: "memory");
#else
  {
    const uint32_t* src = (const uint32_t*)(X + (size_t)rowBase * DD);
    uint32_t* dst = (uint32_t*)As;
#pragma unroll
    for (int i = 0; i < 4; ++i) dst[tid + 256 * i] = src[tid + 256 * i];
  }
#endif
  __syncthreads();
  const int lane = tid & 31, wave = tid >> 5;
  const int colBase = wave * 16;
  const int lcol = lane & 15;
  const int hi = (lane >> 4) & 1;       // half-wave selector (ISA A/B striping)
  v8f acc = {};
#pragma unroll
  for (int c = 0; c < 4; ++c) {          // K = 128 in 4 chunks of 32
    v16bf a, b;
    const bf16_t* ar = As + lcol * DD + c * 32 + hi * 8;
    const bf16_t* wr = W + (size_t)(colBase + lcol) * DD + c * 32 + hi * 8;
#pragma unroll
    for (int e = 0; e < 8; ++e) {
      a[e]     = ar[e];
      a[e + 8] = ar[16 + e];
      b[e]     = wr[e];
      b[e + 8] = wr[16 + e];
    }
    acc = wmma_bf(a, b, acc);
  }
  const int rOff = hi * 8;
  if (BF16OUT) {
    bf16_t* o = (bf16_t*)out;
#pragma unroll
    for (int r = 0; r < 8; ++r)
      o[(size_t)(rowBase + rOff + r) * DD + colBase + lcol] = f2bf(acc[r]);
  } else {
    float* o = (float*)out;
#pragma unroll
    for (int r = 0; r < 8; ++r)
      o[(size_t)(rowBase + rOff + r) * DD + colBase + lcol] = acc[r];
  }
}

// ------------------------------- attention ---------------------------------
// One wave per 16-row Q tile, flash loop over 32-key chunks within the batch.
__global__ void __launch_bounds__(256)
k_attention(const bf16_t* __restrict__ qbf, const bf16_t* __restrict__ kbf,
            const bf16_t* __restrict__ vbf, const float* __restrict__ featsIn,
            float* __restrict__ outF, bf16_t* __restrict__ outB,
            float* __restrict__ resOut) {
  __shared__ bf16_t Pl[8][16 * 32];       // per-wave P staging (C->A transpose)
  const int lane = threadIdx.x & 31, wave = threadIdx.x >> 5;
  const int qtile = blockIdx.x * 8 + wave;
  const int qRow = qtile * 16;
  const int keyBase = (qRow >> 10) << 10;  // batch base (N = 1024)
  const int lcol = lane & 15;
  const int hi = (lane >> 4) & 1;

  v16bf qf[4];
  {
    const bf16_t* qr = qbf + (size_t)(qRow + lcol) * DD + hi * 16;
#pragma unroll
    for (int c = 0; c < 4; ++c)
#pragma unroll
      for (int e = 0; e < 8; ++e) {
        qf[c][e]     = qr[c * 32 + e];
        qf[c][e + 8] = qr[c * 32 + 8 + e];
      }
  }
  v8f o[8];
  float m[8], lsum[8];
  const v8f vzero = {};
#pragma unroll
  for (int i = 0; i < 8; ++i) { o[i] = vzero; m[i] = -1e30f; lsum[i] = 0.0f; }

  for (int kb = 0; kb < Nn; kb += 32) {
    if (kb + 32 < Nn)                      // -> global_prefetch_b8
      __builtin_prefetch(kbf + (size_t)(keyBase + kb + 32) * DD, 0, 1);
    // scores: two 16x16 key tiles
    v8f s0 = vzero, s1 = vzero;
    const bf16_t* kr = kbf + (size_t)(keyBase + kb + lcol) * DD + hi * 16;
#pragma unroll
    for (int c = 0; c < 4; ++c) {
      v16bf b0, b1;
#pragma unroll
      for (int e = 0; e < 16; ++e) {
        b0[e] = kr[c * 32 + e];
        b1[e] = kr[16 * DD + c * 32 + e];
      }
      s0 = wmma_bf(qf[c], b0, s0);
      s1 = wmma_bf(qf[c], b1, s1);
    }
    // online softmax across this 32-key chunk
#pragma unroll
    for (int r = 0; r < 8; ++r) {
      float a0 = s0[r] * NORMF, a1 = s1[r] * NORMF;
      float t = fmaxf(a0, a1);
#pragma unroll
      for (int msk = 8; msk; msk >>= 1) t = fmaxf(t, __shfl_xor(t, msk, 32));
      const float mn = fmaxf(m[r], t);
      const float al = __expf(m[r] - mn);
      const float p0 = __expf(a0 - mn), p1 = __expf(a1 - mn);
      float rs = p0 + p1;
#pragma unroll
      for (int msk = 8; msk; msk >>= 1) rs += __shfl_xor(rs, msk, 32);
      lsum[r] = lsum[r] * al + rs;
      m[r] = mn;
#pragma unroll
      for (int ct = 0; ct < 8; ++ct) o[ct][r] *= al;
      const int prow = hi * 8 + r;
      Pl[wave][prow * 32 + lcol]      = f2bf(p0);
      Pl[wave][prow * 32 + 16 + lcol] = f2bf(p1);
    }
    asm volatile("s_wait_dscnt 0x0" ::: "memory");  // wave-local LDS transpose
    v16bf pa;
    {
      const bf16_t* pr = &Pl[wave][lcol * 32 + hi * 16];
#pragma unroll
      for (int e = 0; e < 8; ++e) { pa[e] = pr[e]; pa[e + 8] = pr[8 + e]; }
    }
#pragma unroll
    for (int ct = 0; ct < 8; ++ct) {
      v16bf bv;
      const bf16_t* vr =
          vbf + (size_t)(keyBase + kb + hi * 16) * DD + ct * 16 + lcol;
#pragma unroll
      for (int e = 0; e < 16; ++e) bv[e] = vr[(size_t)e * DD];
      o[ct] = wmma_bf(pa, bv, o[ct]);
    }
  }
  const int rOff = hi * 8;
#pragma unroll
  for (int r = 0; r < 8; ++r) {
    const float inv = 1.0f / lsum[r];
#pragma unroll
    for (int ct = 0; ct < 8; ++ct) {
      const size_t gi = (size_t)(qRow + rOff + r) * DD + ct * 16 + lcol;
      const float y = o[ct][r] * inv + featsIn[gi];
      outF[gi] = y;
      resOut[gi] = y;
      outB[gi] = f2bf(y);
    }
  }
}

// -------------------- gather -> per-type linear -> scatter -----------------
// One wave per 16-edge tile: gather src rows to LDS (async-to-LDS when
// available), 32 WMMAs vs W (scale = e0/ePerScale for pre/suc), atomic
// scatter-add into temp rows.
__global__ void __launch_bounds__(256)
k_edge_scatter(const bf16_t* __restrict__ featBf, const int* __restrict__ u,
               const int* __restrict__ v, const bf16_t* __restrict__ Wbase,
               int ePerScale, int nEdges, float* __restrict__ temp) {
  __shared__ bf16_t Gl[8][16 * DD];
  const int lane = threadIdx.x & 31, wave = threadIdx.x >> 5;
  const int tile = blockIdx.x * 8 + wave;
  const int e0 = tile * 16;
  if (e0 >= nEdges) return;
  const bf16_t* W = Wbase + (size_t)(e0 / ePerScale) * DD * DD;
#if USE_ASYNC_LDS
  // 16 rows x 8 b128 chunks = 128 async transfers per tile (per-lane LDS addr)
#pragma unroll
  for (int it = 0; it < 4; ++it) {
    const int i = lane + 32 * it;
    const int r = i >> 3;
    const int c = i & 7;
    const int src = v[e0 + r];
    ASYNC_B128(featBf + (size_t)src * DD + c * 8, &Gl[wave][0] + i * 8);
  }
  asm volatile("s_wait_asynccnt 0x0" ::: "memory");
#else
  uint32_t* gld = (uint32_t*)&Gl[wave][0];
  for (int i = lane; i < 16 * 64; i += 32) {          // 16 rows x 64 dwords
    const int r = i >> 6;
    const int c = i & 63;
    const int src = v[e0 + r];
    gld[i] = ((const uint32_t*)(featBf + (size_t)src * DD))[c];
  }
  asm volatile("s_wait_dscnt 0x0" ::: "memory");
#endif
  const int lcol = lane & 15;
  const int hi = (lane >> 4) & 1;
  v8f acc[8];
  const v8f vzero = {};
#pragma unroll
  for (int i = 0; i < 8; ++i) acc[i] = vzero;
#pragma unroll
  for (int c = 0; c < 4; ++c) {
    v16bf a;
    const bf16_t* ar = &Gl[wave][lcol * DD + c * 32 + hi * 16];
#pragma unroll
    for (int e = 0; e < 8; ++e) { a[e] = ar[e]; a[e + 8] = ar[8 + e]; }
#pragma unroll
    for (int ct = 0; ct < 8; ++ct) {
      v16bf b;
      const bf16_t* wr = W + (size_t)(ct * 16 + lcol) * DD + c * 32 + hi * 16;
#pragma unroll
      for (int e = 0; e < 16; ++e) b[e] = wr[e];
      acc[ct] = wmma_bf(a, b, acc[ct]);
    }
  }
  const int rOff = hi * 8;
#pragma unroll
  for (int r = 0; r < 8; ++r) {
    const int dstRow = u[e0 + rOff + r];
    float* trow = temp + (size_t)dstRow * DD + lcol;
#pragma unroll
    for (int ct = 0; ct < 8; ++ct) atomicAdd(trow + ct * 16, acc[ct][r]);
  }
}

// ----------------------- GroupNorm(1,D) + ReLU (bf16 out) ------------------
__global__ void __launch_bounds__(256)
k_gn_relu(const float* __restrict__ in, const float* __restrict__ g,
          const float* __restrict__ bb, bf16_t* __restrict__ outB) {
  const int lane = threadIdx.x & 31, wave = threadIdx.x >> 5;
  const int row = blockIdx.x * 8 + wave;
  const float* x = in + (size_t)row * DD;
  float v4[4]; float s = 0.f;
#pragma unroll
  for (int j = 0; j < 4; ++j) { v4[j] = x[lane + 32 * j]; s += v4[j]; }
  const float mu = waveSum(s) * (1.0f / DD);
  float d = 0.f;
#pragma unroll
  for (int j = 0; j < 4; ++j) { const float t = v4[j] - mu; d += t * t; }
  const float rstd = rsqrtf(waveSum(d) * (1.0f / DD) + EPSF);
#pragma unroll
  for (int j = 0; j < 4; ++j) {
    const int c = lane + 32 * j;
    const float y = fmaxf((v4[j] - mu) * rstd * g[c] + bb[c], 0.f);
    outB[(size_t)row * DD + c] = f2bf(y);
  }
}

// --------------- GroupNorm (no act) + residual-ReLU, res update ------------
__global__ void __launch_bounds__(256)
k_gn2_res(const float* __restrict__ in, const float* __restrict__ g,
          const float* __restrict__ bb, float* __restrict__ res,
          float* __restrict__ outF, bf16_t* __restrict__ outB) {
  const int lane = threadIdx.x & 31, wave = threadIdx.x >> 5;
  const int row = blockIdx.x * 8 + wave;
  const float* x = in + (size_t)row * DD;
  float v4[4]; float s = 0.f;
#pragma unroll
  for (int j = 0; j < 4; ++j) { v4[j] = x[lane + 32 * j]; s += v4[j]; }
  const float mu = waveSum(s) * (1.0f / DD);
  float d = 0.f;
#pragma unroll
  for (int j = 0; j < 4; ++j) { const float t = v4[j] - mu; d += t * t; }
  const float rstd = rsqrtf(waveSum(d) * (1.0f / DD) + EPSF);
#pragma unroll
  for (int j = 0; j < 4; ++j) {
    const int c = lane + 32 * j;
    const size_t gi = (size_t)row * DD + c;
    const float t = (v4[j] - mu) * rstd * g[c] + bb[c];
    const float y = fmaxf(t + res[gi], 0.f);
    res[gi] = y;
    outF[gi] = y;
    outB[gi] = f2bf(y);
  }
}

// ---------------------------------------------------------------------------
extern "C" void kernel_launch(void* const* d_in, const int* in_sizes, int n_in,
                              void* d_out, int out_size, void* d_ws,
                              size_t ws_size, hipStream_t stream) {
  (void)in_sizes; (void)n_in; (void)out_size; (void)ws_size;
  const float* ctrs    = (const float*)d_in[0];
  const float* feats   = (const float*)d_in[1];
  const float* Wq      = (const float*)d_in[2];
  const float* Wk      = (const float*)d_in[3];
  const float* Wv      = (const float*)d_in[4];
  const float* W_ctr   = (const float*)d_in[5];
  const float* W_pre   = (const float*)d_in[6];
  const float* W_suc   = (const float*)d_in[7];
  const float* W_left  = (const float*)d_in[8];
  const float* W_right = (const float*)d_in[9];
  const float* W_ctr2  = (const float*)d_in[10];
  const float* gn_g    = (const float*)d_in[11];
  const float* gn_b    = (const float*)d_in[12];
  const float* gn2_g   = (const float*)d_in[13];
  const float* gn2_b   = (const float*)d_in[14];
  const int* pre_u   = (const int*)d_in[15];
  const int* pre_v   = (const int*)d_in[16];
  const int* suc_u   = (const int*)d_in[17];
  const int* suc_v   = (const int*)d_in[18];
  const int* left_u  = (const int*)d_in[19];
  const int* left_v  = (const int*)d_in[20];
  const int* right_u = (const int*)d_in[21];
  const int* right_v = (const int*)d_in[22];

  // workspace carve (~38 MB), 256B aligned
  char* p = (char*)d_ws;
  auto carve = [&](size_t bytes) -> void* {
    void* r = (void*)p; p += (bytes + 255) & ~(size_t)255; return r;
  };
  const size_t MD = (size_t)Mrows * DD;
  const size_t WW = (size_t)DD * DD;
  float*  res      = (float*)carve(MD * 4);
  float*  temp     = (float*)carve(MD * 4);
  bf16_t* ctrs_bf  = (bf16_t*)carve(MD * 2);
  bf16_t* feats_bf = (bf16_t*)carve(MD * 2);  // reused as feat_bf after attn
  bf16_t* q_bf     = (bf16_t*)carve(MD * 2);
  bf16_t* k_bf     = (bf16_t*)carve(MD * 2);
  bf16_t* v_bf     = (bf16_t*)carve(MD * 2);
  bf16_t* Wq_bf    = (bf16_t*)carve(WW * 2);
  bf16_t* Wk_bf    = (bf16_t*)carve(WW * 2);
  bf16_t* Wv_bf    = (bf16_t*)carve(WW * 2);
  bf16_t* Wctr_bf  = (bf16_t*)carve(WW * Ll * 2);
  bf16_t* Wpre_bf  = (bf16_t*)carve(WW * Ll * Ss * 2);
  bf16_t* Wsuc_bf  = (bf16_t*)carve(WW * Ll * Ss * 2);
  bf16_t* Wleft_bf = (bf16_t*)carve(WW * Ll * 2);
  bf16_t* Wright_bf= (bf16_t*)carve(WW * Ll * 2);
  bf16_t* Wctr2_bf = (bf16_t*)carve(WW * Ll * 2);
  bf16_t* feat_bf  = feats_bf;           // alias: feats_bf dead after K/V gemm
  float*  feat     = (float*)d_out;      // f32 path only needed for output

  auto cvt = [&](const float* s, bf16_t* dst, size_t n) {
    k_f32_to_bf16<<<(unsigned)((n + 255) / 256), 256, 0, stream>>>(s, dst,
                                                                   (int)n);
  };
  cvt(ctrs, ctrs_bf, MD);
  cvt(feats, feats_bf, MD);
  cvt(Wq, Wq_bf, WW);
  cvt(Wk, Wk_bf, WW);
  cvt(Wv, Wv_bf, WW);
  cvt(W_ctr, Wctr_bf, WW * Ll);
  cvt(W_pre, Wpre_bf, WW * Ll * Ss);
  cvt(W_suc, Wsuc_bf, WW * Ll * Ss);
  cvt(W_left, Wleft_bf, WW * Ll);
  cvt(W_right, Wright_bf, WW * Ll);
  cvt(W_ctr2, Wctr2_bf, WW * Ll);

  // QKV projections (bf16 out for the attention WMMAs)
  k_gemm_xwT<true><<<Mrows / 16, 256, 0, stream>>>(ctrs_bf, Wq_bf, q_bf);
  k_gemm_xwT<true><<<Mrows / 16, 256, 0, stream>>>(feats_bf, Wk_bf, k_bf);
  k_gemm_xwT<true><<<Mrows / 16, 256, 0, stream>>>(feats_bf, Wv_bf, v_bf);

  // feat = feats + softmax(q k^T / sqrt(D)) v ; also seeds res and feat_bf
  k_attention<<<Mrows / 16 / 8, 256, 0, stream>>>(q_bf, k_bf, v_bf, feats,
                                                  feat, feat_bf, res);

  const int nPS = Ss * Ee;               // 196608 pre/suc edges per layer
  for (int i = 0; i < Ll; ++i) {
    k_gemm_xwT<false><<<Mrows / 16, 256, 0, stream>>>(
        feat_bf, Wctr_bf + (size_t)i * WW, temp);
    k_edge_scatter<<<(nPS / 16 + 7) / 8, 256, 0, stream>>>(
        feat_bf, pre_u, pre_v, Wpre_bf + (size_t)i * Ss * WW, Ee, nPS, temp);
    k_edge_scatter<<<(nPS / 16 + 7) / 8, 256, 0, stream>>>(
        feat_bf, suc_u, suc_v, Wsuc_bf + (size_t)i * Ss * WW, Ee, nPS, temp);
    k_edge_scatter<<<(Ee / 16 + 7) / 8, 256, 0, stream>>>(
        feat_bf, left_u, left_v, Wleft_bf + (size_t)i * WW, Ee, Ee, temp);
    k_edge_scatter<<<(Ee / 16 + 7) / 8, 256, 0, stream>>>(
        feat_bf, right_u, right_v, Wright_bf + (size_t)i * WW, Ee, Ee, temp);
    k_gn_relu<<<Mrows / 8, 256, 0, stream>>>(temp, gn_g + i * DD,
                                             gn_b + i * DD, feat_bf);
    k_gemm_xwT<false><<<Mrows / 16, 256, 0, stream>>>(
        feat_bf, Wctr2_bf + (size_t)i * WW, temp);
    k_gn2_res<<<Mrows / 8, 256, 0, stream>>>(temp, gn2_g + i * DD,
                                             gn2_b + i * DD, res, feat,
                                             feat_bf);
  }
  // final feat already written to d_out by the last k_gn2_res
}
